// MPMLP_21182778704945
// MI455X (gfx1250) — matrix operand order repaired and checked by reading, not compiled, not run on hardware
//
#include <hip/hip_runtime.h>

// ---------------------------------------------------------------------------
// MPMLP: y = gelu(x * w1)  (per-channel, MULTI copies)
//        z = gelu(Wg[m] @ y[:,m])  (4 dense 64x64 channel matmuls)
//        out[c] = sum_m z[m,c] * w2[c,m]
// Strategy: 16-pixel tiles; one wave per group m; Wg[m] resident in VGPRs as
// WMMA B-tiles (f16); v_wmma_f32_16x16x32_f16 with f32 accumulation; LDS only
// for the tiny cross-m reduction.
// ---------------------------------------------------------------------------

typedef _Float16 v16h __attribute__((ext_vector_type(16)));
typedef float    v8f  __attribute__((ext_vector_type(8)));

#define DIM    64
#define MULTI  4
#define NPIX   (8 * 256 * 256)
#define TILE   16
#define NT     (NPIX / TILE)   // 32768 pixel tiles
#define NBLK   2048
#define TPB    128             // 4 wave32s

// tanh-approx GELU via exp2/rcp (2 transcendentals + 4 VALU):
// gelu(v) = v * sigmoid(1.5957691*(v + 0.044715 v^3))
//         = v / (1 + exp2(v * (-2.3021181 - 0.1029442 v^2)))
__device__ __forceinline__ float fast_gelu(float v) {
    float v2 = v * v;
    float u  = v * __builtin_fmaf(-0.1029442127f, v2, -2.3021180685f);
    float e  = __builtin_amdgcn_exp2f(u);
    return v * __builtin_amdgcn_rcpf(1.0f + e);
}

__global__ __launch_bounds__(TPB) void mpmlp_kernel(
    const float* __restrict__ x,  const float* __restrict__ w1,
    const float* __restrict__ wg, const float* __restrict__ w2,
    float* __restrict__ out)
{
    __shared__ float zslab[MULTI * TILE * DIM];   // 4 waves x 16 pix x 64 ch = 16 KB

    const int tid  = threadIdx.x;
    const int m    = tid >> 5;      // wave id == group index m
    const int lane = tid & 31;
    const int l    = lane & 15;     // N coordinate (B/C layout) / pixel row (A layout)
    const int h    = lane >> 4;     // lane-half selects K sub-ranges

    // ---- preload Wg[m] as f16 WMMA B-tiles: B[n][k] covers N=16n..16n+15,
    //      K(=input chan) = 32k+8h+j (j<8), 32k+16+8h+(j-8) (j>=8) per lane.
    v16h Bm[4][2];
    #pragma unroll
    for (int n = 0; n < 4; ++n) {
        #pragma unroll
        for (int k = 0; k < 2; ++k) {
            const float* wrow = wg + ((size_t)(m * DIM + 16 * n + l)) * DIM + 32 * k + 8 * h;
            const float4* w4 = (const float4*)wrow;
            #pragma unroll
            for (int q = 0; q < 4; ++q) {
                float4 v = w4[q + 2 * (q >> 1)];
                Bm[n][k][4 * q + 0] = (_Float16)v.x;
                Bm[n][k][4 * q + 1] = (_Float16)v.y;
                Bm[n][k][4 * q + 2] = (_Float16)v.z;
                Bm[n][k][4 * q + 3] = (_Float16)v.w;
            }
        }
    }

    // ---- preload w1[:,m] for this lane's 32 channels.
    // A-layout channels per lane: run (c,s): ch = 32c + 16s + 8h + j, j=0..7
    float w1r[4][8];
    #pragma unroll
    for (int c = 0; c < 2; ++c) {
        #pragma unroll
        for (int s = 0; s < 2; ++s) {
            int ch0 = 32 * c + 16 * s + 8 * h;
            #pragma unroll
            for (int j = 0; j < 8; ++j)
                w1r[2 * c + s][j] = w1[(ch0 + j) * MULTI + m];
        }
    }

    // ---- preload w2[:,m] for this lane's 4 output-column tiles
    float w2r[4];
    #pragma unroll
    for (int n = 0; n < 4; ++n)
        w2r[n] = w2[(16 * n + l) * MULTI + m];

    for (int t = blockIdx.x; t < NT; t += NBLK) {
        const float* xpix = x + ((size_t)(t * TILE + l)) * DIM;   // pixel row M = l
        if (t + NBLK < NT)  // uniform per block -> EXEC stays all-ones
            __builtin_prefetch(x + ((size_t)((t + NBLK) * TILE + l)) * DIM, 0, 0);

        // ---- build A tiles (y in f16, A-layout): A[c].h[8s+j] = y(ch=32c+16s+8h+j)
        v16h A[2];
        #pragma unroll
        for (int c = 0; c < 2; ++c) {
            #pragma unroll
            for (int s = 0; s < 2; ++s) {
                int ch0 = 32 * c + 16 * s + 8 * h;
                const float4* x4 = (const float4*)(xpix + ch0);
                float4 v0 = x4[0], v1 = x4[1];
                float xs[8] = {v0.x, v0.y, v0.z, v0.w, v1.x, v1.y, v1.z, v1.w};
                #pragma unroll
                for (int j = 0; j < 8; ++j) {
                    float y = fast_gelu(xs[j] * w1r[2 * c + s][j]);
                    A[c][8 * s + j] = (_Float16)y;
                }
            }
        }

        // ---- z[m] = Wg[m] @ y[:,m] : 8 WMMAs (4 col tiles x K-split 2), f32 accum
        v8f acc[4];
        #pragma unroll
        for (int n = 0; n < 4; ++n) {
            v8f cacc = {0.f, 0.f, 0.f, 0.f, 0.f, 0.f, 0.f, 0.f};
            cacc = __builtin_amdgcn_wmma_f32_16x16x32_f16(
                false, A[0], false, Bm[n][0], (short)0, cacc, false, false);
            cacc = __builtin_amdgcn_wmma_f32_16x16x32_f16(
                false, A[1], false, Bm[n][1], (short)0, cacc, false, false);
            acc[n] = cacc;
        }

        // ---- gelu(z) * w2  ->  LDS slab (C layout: VGPR r = pixel r+8h, N = l)
        #pragma unroll
        for (int n = 0; n < 4; ++n) {
            #pragma unroll
            for (int r = 0; r < 8; ++r) {
                float g = fast_gelu(acc[n][r]) * w2r[n];
                zslab[((m * 4 + n) * 8 + r) * 32 + lane] = g;  // conflict-free
            }
        }

        __syncthreads();

        // ---- cross-m reduction + coalesced store: thread -> 8 consecutive outs
        {
            int f0 = tid * 8;            // flat (pixel*64 + chan) within tile
            int p  = f0 >> 6;            // pixel 0..15
            int c0 = f0 & 63;            // chan base (multiple of 8 -> single n)
            int n  = c0 >> 4;
            int r  = p & 7;
            int hh = p >> 3;
            float o[8];
            #pragma unroll
            for (int j = 0; j < 8; ++j) {
                int L2 = 16 * hh + ((c0 + j) & 15);
                float ssum = 0.f;
                #pragma unroll
                for (int w = 0; w < 4; ++w)
                    ssum += zslab[((w * 4 + n) * 8 + r) * 32 + L2];
                o[j] = ssum;
            }
            float* op = out + (size_t)t * (TILE * DIM) + f0;
            float4 s0 = {o[0], o[1], o[2], o[3]};
            float4 s1 = {o[4], o[5], o[6], o[7]};
            ((float4*)op)[0] = s0;
            ((float4*)op)[1] = s1;
        }
        __syncthreads();   // zslab reused next iteration
    }
}

extern "C" void kernel_launch(void* const* d_in, const int* in_sizes, int n_in,
                              void* d_out, int out_size, void* d_ws, size_t ws_size,
                              hipStream_t stream) {
    const float* x  = (const float*)d_in[0];
    const float* w1 = (const float*)d_in[1];
    const float* wg = (const float*)d_in[2];
    const float* w2 = (const float*)d_in[3];
    float* out = (float*)d_out;
    hipLaunchKernelGGL(mpmlp_kernel, dim3(NBLK), dim3(TPB), 0, stream,
                       x, w1, wg, w2, out);
}
